// MultiHeadAttention_17214228922464
// MI455X (gfx1250) — matrix operand order repaired and checked by reading, not compiled, not run on hardware
//
#include <hip/hip_runtime.h>
#include <hip/hip_bf16.h>

typedef _Float16 half_t;
typedef __attribute__((ext_vector_type(16))) _Float16 v16h;
typedef __attribute__((ext_vector_type(8)))  float    v8f;
typedef __attribute__((ext_vector_type(4)))  unsigned u32x4;

constexpr int BB   = 4;
constexpr int NSEQ = 2048;
constexpr int EMB  = 512;
constexpr int NH   = 8;
constexpr int DH   = 64;
constexpr int E3   = 3 * EMB;
constexpr int TOK  = BB * NSEQ;            // 8192
constexpr float SCALE = 0.125f;            // 1/sqrt(64)

__device__ __forceinline__ v8f wmma16(v16h a, v16h b, v8f c) {
    // D = A(16x32 f16) * B(32x16 f16) + C(16x16 f32)
    return __builtin_amdgcn_wmma_f32_16x16x32_f16(false, a, false, b, (short)0, c,
                                                  false, false);
}

// ---- A-fragment (16x32, f16). Row-major source, K contiguous, p0 -> (row0,k0).
// ISA layout: lane r(=lane&15) holds row r; lanes 16-31 hold K+8 within each
// 16-chunk; VGPRs 0-3 cover K 0..7(+8), VGPRs 4-7 cover K 16..23(+8).
__device__ __forceinline__ v16h load_frag_a(const half_t* p0, int ld) {
    const int lane = threadIdx.x & 31;
    const int r    = lane & 15;
    const int hs   = lane >> 4;
    const half_t* p = p0 + r * ld + hs * 8;
    union { v16h h; unsigned u[8]; } f;
#pragma unroll
    for (int v = 0; v < 8; ++v) {
        const int koff = ((v & 4) << 2) + ((v & 3) << 1);  // {0,2,4,6,16,18,20,22}
        f.u[v] = *(const unsigned*)(p + koff);
    }
    return f.h;
}

// ---- B-fragment (32x16, f16) loaded from Bt (16 rows = output cols, K contig).
// ISA layout: same K order as A: lanes 16-31 hold K+8 within each 16-chunk;
// elements 0-7 cover K 0..7(+8), elements 8-15 cover K 16..23(+8).
__device__ __forceinline__ v16h load_frag_b(const half_t* p0, int ld) {
    const int lane = threadIdx.x & 31;
    const int r    = lane & 15;
    const int hs   = lane >> 4;
    const half_t* p = p0 + r * ld + hs * 8;
    union { v16h h; u32x4 q[2]; } f;
    f.q[0] = *(const u32x4*)(p);
    f.q[1] = *(const u32x4*)(p + 16);
    return f.h;
}

// ---------------- fp32 -> fp16 convert ----------------
__global__ void cvt_f32_f16_kernel(const float* __restrict__ src,
                                   half_t* __restrict__ dst, int n) {
    int i = blockIdx.x * blockDim.x + threadIdx.x;
    if (i < n) dst[i] = (half_t)src[i];
}

// ---------------- QKV GEMM: qkv[n][e3] = sum_c xh[n][c] * w_qkv[e3][c] -------
// Block: 256 thr (8 waves), 128x128 tile; wave = 64x32 (4x2 wmma tiles).
// Epilogue scatters to q/k/v in [b][h][n][d] layout; SCALE folded into Q.
__global__ __launch_bounds__(256) void qkv_gemm_kernel(
        const half_t* __restrict__ xh, const half_t* __restrict__ wq,
        half_t* __restrict__ qh, half_t* __restrict__ kh, half_t* __restrict__ vh) {
    const int lane = threadIdx.x & 31;
    const int wid  = threadIdx.x >> 5;
    const int m0 = blockIdx.x * 128 + (wid & 1) * 64;
    const int n0 = blockIdx.y * 128 + (wid >> 1) * 32;

    v8f acc[4][2];
#pragma unroll
    for (int im = 0; im < 4; ++im)
#pragma unroll
        for (int in = 0; in < 2; ++in)
            acc[im][in] = (v8f){0.f,0.f,0.f,0.f,0.f,0.f,0.f,0.f};

    for (int k0 = 0; k0 < EMB; k0 += 32) {
        v16h a[4], b[2];
#pragma unroll
        for (int im = 0; im < 4; ++im)
            a[im] = load_frag_a(xh + (size_t)(m0 + 16 * im) * EMB + k0, EMB);
#pragma unroll
        for (int in = 0; in < 2; ++in)
            b[in] = load_frag_b(wq + (size_t)(n0 + 16 * in) * EMB + k0, EMB);
#pragma unroll
        for (int im = 0; im < 4; ++im)
#pragma unroll
            for (int in = 0; in < 2; ++in)
                acc[im][in] = wmma16(a[im], b[in], acc[im][in]);
    }

    const int col = lane & 15, hs = lane >> 4;
#pragma unroll
    for (int im = 0; im < 4; ++im)
#pragma unroll
        for (int in = 0; in < 2; ++in)
#pragma unroll
            for (int i = 0; i < 8; ++i) {
                const int row = m0 + im * 16 + hs * 8 + i;
                const int e3  = n0 + in * 16 + col;
                const int s   = e3 >> 9;          // 0:q 1:k 2:v (tile-uniform)
                const int hd  = e3 & 511;
                const int b_  = row >> 11, n_ = row & (NSEQ - 1);
                const size_t di = ((size_t)(b_ * NH + (hd >> 6)) * NSEQ + n_) * DH + (hd & 63);
                const float v = acc[im][in][i];
                if (s == 0)      qh[di] = (half_t)(v * SCALE);
                else if (s == 1) kh[di] = (half_t)v;
                else             vh[di] = (half_t)v;
            }
}

// ---------------- Flash attention: one block = 128 q-rows of one (b,h) -------
__global__ __launch_bounds__(256) void attn_kernel(
        const half_t* __restrict__ qh, const half_t* __restrict__ kh,
        const half_t* __restrict__ vh, half_t* __restrict__ oh) {
    __shared__ half_t Kt[32 * 64];        // [key][d], d contiguous
    __shared__ half_t Vt[64 * 32];        // [d][key], key contiguous (transposed)
    __shared__ half_t Ps[8 * 16 * 32];    // per-wave P staging, [qrow][key]

    const int tid = threadIdx.x, lane = tid & 31, wid = tid >> 5;
    const int bh = blockIdx.y;                // b*H + h
    const int b_ = bh >> 3, h_ = bh & 7;
    const int q0 = blockIdx.x * 128 + wid * 16;
    const int col = lane & 15, hs = lane >> 4;

    const half_t* qbase = qh + ((size_t)bh * NSEQ + q0) * DH;
    const v16h qa0 = load_frag_a(qbase, DH);       // K = d 0..31
    const v16h qa1 = load_frag_a(qbase + 32, DH);  // K = d 32..63

    v8f o[4];
#pragma unroll
    for (int t = 0; t < 4; ++t) o[t] = (v8f){0.f,0.f,0.f,0.f,0.f,0.f,0.f,0.f};
    float mrun[8], lrun[8];
#pragma unroll
    for (int i = 0; i < 8; ++i) { mrun[i] = -1e30f; lrun[i] = 0.f; }

    const half_t* kbh = kh + (size_t)bh * NSEQ * DH;
    const half_t* vbh = vh + (size_t)bh * NSEQ * DH;
    half_t* pw = Ps + wid * (16 * 32);

    for (int j0 = 0; j0 < NSEQ; j0 += 32) {
        // ---- cooperative K/V tile load (2048 halfs each, 8 per thread)
        const half_t* ks = kbh + (size_t)j0 * DH;
        *(u32x4*)(Kt + tid * 8) = *(const u32x4*)(ks + tid * 8);
        const half_t* vs = vbh + (size_t)j0 * DH;
        {
            const int m = tid >> 3, d0 = (tid & 7) * 8;
            u32x4 vv = *(const u32x4*)(vs + m * DH + d0);
            const half_t* hv = (const half_t*)&vv;
#pragma unroll
            for (int t = 0; t < 8; ++t) Vt[(d0 + t) * 32 + m] = hv[t];
        }
        if (j0 + 32 < NSEQ) {
            __builtin_prefetch(ks + 32 * DH + tid * 8, 0, 0);
            __builtin_prefetch(vs + 32 * DH + tid * 8, 0, 0);
        }
        __syncthreads();

        // ---- S = Q * K^T   (16 q-rows x 32 keys, scale pre-folded into Q)
        v8f s0 = (v8f){0.f,0.f,0.f,0.f,0.f,0.f,0.f,0.f};
        v8f s1 = (v8f){0.f,0.f,0.f,0.f,0.f,0.f,0.f,0.f};
        s0 = wmma16(qa0, load_frag_b(Kt + 0 * DH + 0,  DH), s0);
        s0 = wmma16(qa1, load_frag_b(Kt + 0 * DH + 32, DH), s0);
        s1 = wmma16(qa0, load_frag_b(Kt + 16 * DH + 0,  DH), s1);
        s1 = wmma16(qa1, load_frag_b(Kt + 16 * DH + 32, DH), s1);

        // ---- online softmax (rows live as VGPR idx + lane-half; cols = lanes)
        float mb[8];
#pragma unroll
        for (int i = 0; i < 8; ++i) mb[i] = fmaxf(s0[i], s1[i]);
#pragma unroll
        for (int off = 1; off < 16; off <<= 1)
#pragma unroll
            for (int i = 0; i < 8; ++i)
                mb[i] = fmaxf(mb[i], __shfl_xor(mb[i], off, 32));
        float alpha[8];
#pragma unroll
        for (int i = 0; i < 8; ++i) {
            const float mn = fmaxf(mrun[i], mb[i]);
            alpha[i] = __expf(mrun[i] - mn);
            mrun[i]  = mn;
        }
#pragma unroll
        for (int i = 0; i < 8; ++i) {
            s0[i] = __expf(s0[i] - mrun[i]);
            s1[i] = __expf(s1[i] - mrun[i]);
        }
        float ls[8];
#pragma unroll
        for (int i = 0; i < 8; ++i) ls[i] = s0[i] + s1[i];
#pragma unroll
        for (int off = 1; off < 16; off <<= 1)
#pragma unroll
            for (int i = 0; i < 8; ++i) ls[i] += __shfl_xor(ls[i], off, 32);
#pragma unroll
        for (int i = 0; i < 8; ++i) lrun[i] = lrun[i] * alpha[i] + ls[i];
#pragma unroll
        for (int t = 0; t < 4; ++t)
#pragma unroll
            for (int i = 0; i < 8; ++i) o[t][i] *= alpha[i];

        // ---- P: C-layout -> A-layout via per-wave LDS round-trip
#pragma unroll
        for (int i = 0; i < 8; ++i) {
            pw[(hs * 8 + i) * 32 + col]      = (half_t)s0[i];
            pw[(hs * 8 + i) * 32 + 16 + col] = (half_t)s1[i];
        }
        const v16h pa = load_frag_a(pw, 32);   // same-wave LDS RAW: DS in-order

        // ---- O += P * V
#pragma unroll
        for (int t = 0; t < 4; ++t)
            o[t] = wmma16(pa, load_frag_b(Vt + t * 16 * 32, 32), o[t]);

        __syncthreads();
    }

    // ---- normalize + store to oh[b][n][h*64+d]
#pragma unroll
    for (int i = 0; i < 8; ++i) {
        const float rl = 1.0f / lrun[i];
        const int row = q0 + hs * 8 + i;
        half_t* orow = oh + ((size_t)b_ * NSEQ + row) * EMB + h_ * DH;
#pragma unroll
        for (int t = 0; t < 4; ++t)
            orow[t * 16 + col] = (half_t)(o[t][i] * rl);
    }
}

// ---------------- Output proj: out[n][e] = sum_c oh[n][c]*w_proj[e][c] + b[e] -
__global__ __launch_bounds__(256) void proj_gemm_kernel(
        const half_t* __restrict__ ohh, const half_t* __restrict__ wp,
        const float* __restrict__ bias, float* __restrict__ out) {
    const int lane = threadIdx.x & 31;
    const int wid  = threadIdx.x >> 5;
    const int m0 = blockIdx.x * 128 + (wid & 1) * 64;
    const int n0 = blockIdx.y * 128 + (wid >> 1) * 32;

    v8f acc[4][2];
#pragma unroll
    for (int im = 0; im < 4; ++im)
#pragma unroll
        for (int in = 0; in < 2; ++in)
            acc[im][in] = (v8f){0.f,0.f,0.f,0.f,0.f,0.f,0.f,0.f};

    for (int k0 = 0; k0 < EMB; k0 += 32) {
        v16h a[4], b[2];
#pragma unroll
        for (int im = 0; im < 4; ++im)
            a[im] = load_frag_a(ohh + (size_t)(m0 + 16 * im) * EMB + k0, EMB);
#pragma unroll
        for (int in = 0; in < 2; ++in)
            b[in] = load_frag_b(wp + (size_t)(n0 + 16 * in) * EMB + k0, EMB);
#pragma unroll
        for (int im = 0; im < 4; ++im)
#pragma unroll
            for (int in = 0; in < 2; ++in)
                acc[im][in] = wmma16(a[im], b[in], acc[im][in]);
    }

    const int col = lane & 15, hs = lane >> 4;
#pragma unroll
    for (int im = 0; im < 4; ++im)
#pragma unroll
        for (int in = 0; in < 2; ++in) {
            const int e = n0 + in * 16 + col;
            const float bb = bias[e];
#pragma unroll
            for (int i = 0; i < 8; ++i) {
                const int row = m0 + im * 16 + hs * 8 + i;
                out[(size_t)row * EMB + e] = acc[im][in][i] + bb;
            }
        }
}

extern "C" void kernel_launch(void* const* d_in, const int* in_sizes, int n_in,
                              void* d_out, int out_size, void* d_ws, size_t ws_size,
                              hipStream_t stream) {
    const float* x      = (const float*)d_in[0];
    const float* w_qkv  = (const float*)d_in[1];
    const float* w_proj = (const float*)d_in[2];
    const float* b_proj = (const float*)d_in[3];
    float* out = (float*)d_out;

    half_t* xh  = (half_t*)d_ws;                 // TOK*EMB
    half_t* wqh = xh  + (size_t)TOK * EMB;       // E3*EMB
    half_t* wph = wqh + (size_t)E3  * EMB;       // EMB*EMB
    half_t* qh  = wph + (size_t)EMB * EMB;       // TOK*EMB  ([b][h][n][d])
    half_t* kh  = qh  + (size_t)TOK * EMB;
    half_t* vh  = kh  + (size_t)TOK * EMB;
    half_t* ohh = vh  + (size_t)TOK * EMB;       // TOK*EMB  ([b][n][h*d])

    cvt_f32_f16_kernel<<<dim3((TOK * EMB + 255) / 256), 256, 0, stream>>>(x, xh, TOK * EMB);
    cvt_f32_f16_kernel<<<dim3((E3 * EMB + 255) / 256), 256, 0, stream>>>(w_qkv, wqh, E3 * EMB);
    cvt_f32_f16_kernel<<<dim3((EMB * EMB + 255) / 256), 256, 0, stream>>>(w_proj, wph, EMB * EMB);

    qkv_gemm_kernel<<<dim3(TOK / 128, E3 / 128), 256, 0, stream>>>(xh, wqh, qh, kh, vh);
    attn_kernel<<<dim3(NSEQ / 128, BB * NH), 256, 0, stream>>>(qh, kh, vh, ohh);
    proj_gemm_kernel<<<dim3(TOK / 128, EMB / 128), 256, 0, stream>>>(ohh, wph, b_proj, out);
}
